// NeuronMistral3DecoderLayer_11089605558853
// MI455X (gfx1250) — compile-verified, matrix operand order checked by reading, not run on hardware
//
#include <hip/hip_runtime.h>
#include <hip/hip_bf16.h>

// ---------------------------------------------------------------------------
// Mistral decoder layer for gfx1250 (MI455X).
// fp32 I/O; bf16 WMMA compute; async global->LDS double-buffered GEMM.
// ---------------------------------------------------------------------------

#define S_LEN 2048
#define HID   4096
#define NH    32
#define NKV   8
#define HD    128
#define KVD   (NKV * HD)
#define FF    14336
#define EPSV  1e-5f

typedef __bf16 bf16_t;
typedef bf16_t v16bf __attribute__((ext_vector_type(16)));
typedef float  v8f   __attribute__((ext_vector_type(8)));
typedef int    gv4i  __attribute__((vector_size(16)));   // matches async builtin pointee

union FragBF {
    unsigned u[8];
    v16bf    v;
};

__device__ __forceinline__ unsigned short f2bf(float f) {
    unsigned u = __builtin_bit_cast(unsigned, f);
    unsigned r = u + 0x7FFFu + ((u >> 16) & 1u);   // round-to-nearest-even
    return (unsigned short)(r >> 16);
}
__device__ __forceinline__ float bf2f(unsigned short h) {
    return __builtin_bit_cast(float, ((unsigned)h) << 16);
}
__device__ __forceinline__ unsigned pack2(float a, float b) {
    return (unsigned)f2bf(a) | ((unsigned)f2bf(b) << 16);
}

// intra-wave LDS producer->consumer ordering (CDNA5 split DS counter)
__device__ __forceinline__ void wave_lds_sync() {
    asm volatile("s_wait_dscnt 0" ::: "memory");
    __builtin_amdgcn_wave_barrier();
}

// fragment K-pair index: kb = (v<4 ? 2v : 16+2(v-4)) + 8*half  ->  kp = kb/2
__device__ __forceinline__ int frag_kp(int v, int half) {
    return (v < 4 ? v : v + 4) + 4 * half;
}

// ---------------------------------------------------------------------------
// CDNA5 async global->LDS copy (ASYNCcnt) with synchronous fallback
// ---------------------------------------------------------------------------
#if defined(__has_builtin)
#if __has_builtin(__builtin_amdgcn_global_load_async_to_lds_b128)
#define HAVE_ASYNC_LDS 1
#endif
#endif
#ifndef HAVE_ASYNC_LDS
#define HAVE_ASYNC_LDS 0
#endif

__device__ __forceinline__ void async_copy16(const unsigned short* g, unsigned short* l) {
#if HAVE_ASYNC_LDS
    __builtin_amdgcn_global_load_async_to_lds_b128(
        (__attribute__((address_space(1))) gv4i*)(unsigned long long)g,
        (__attribute__((address_space(3))) gv4i*)(unsigned)(unsigned long long)l,
        0, 0);
#else
    *(uint4*)l = *(const uint4*)g;
#endif
}

template <int N>
__device__ __forceinline__ void wait_async() {
#if HAVE_ASYNC_LDS
#if __has_builtin(__builtin_amdgcn_s_wait_asynccnt)
    __builtin_amdgcn_s_wait_asynccnt(N);
#else
    asm volatile("s_wait_asynccnt %0" ::"i"(N) : "memory");
#endif
#endif
}

// ---------------------------------------------------------------------------
// RMSNorm: one block per row, bf16 output
// ---------------------------------------------------------------------------
__global__ __launch_bounds__(256) void rmsnorm_kernel(const float* __restrict__ x,
                                                      const float* __restrict__ w,
                                                      unsigned short* __restrict__ out) {
    const int row = blockIdx.x;
    const int tid = threadIdx.x;
    const float* xr = x + (size_t)row * HID;
    float ss = 0.f;
    for (int i = tid; i < HID; i += 256) { float v = xr[i]; ss += v * v; }
    __shared__ float red[256];
    red[tid] = ss;
    __syncthreads();
    for (int s = 128; s > 0; s >>= 1) {
        if (tid < s) red[tid] += red[tid + s];
        __syncthreads();
    }
    const float inv = rsqrtf(red[0] / (float)HID + EPSV);
    for (int i = tid; i < HID; i += 256)
        out[(size_t)row * HID + i] = f2bf(xr[i] * inv * w[i]);
}

// ---------------------------------------------------------------------------
// Weight convert + transpose: fp32 [K,N] row-major -> bf16 [N,K] row-major
// ---------------------------------------------------------------------------
__global__ __launch_bounds__(256) void wcvt_t_kernel(const float* __restrict__ in,
                                                     unsigned short* __restrict__ out,
                                                     int K, int N) {
    __shared__ float tile[32][33];
    const int nb = blockIdx.x * 32, kb = blockIdx.y * 32;
    const int tx = threadIdx.x & 31, ty = threadIdx.x >> 5;   // ty = 0..7
    for (int i = ty; i < 32; i += 8)
        tile[i][tx] = in[(size_t)(kb + i) * N + nb + tx];
    __syncthreads();
    for (int i = ty; i < 32; i += 8)
        out[(size_t)(nb + i) * K + kb + tx] = f2bf(tile[tx][i]);
}

// ---------------------------------------------------------------------------
// Tiled GEMM: C[M,N] = A[M,K] @ Bt[N,K]^T   (both bf16)
// Double-buffered async global->LDS staging; v_wmma_f32_16x16x32_bf16.
// Block tile 128x128, 8 waves (4x2), wave tile 32x64, K step 32.
// MODE: 0 = f32 out, 1 = f32 out + residual R, 2 = bf16 out.
// ---------------------------------------------------------------------------
template <int MODE>
__global__ __launch_bounds__(256) void gemm_bf16_wmma(const unsigned short* __restrict__ A,
                                                      const unsigned short* __restrict__ Bt,
                                                      const float* __restrict__ R,
                                                      float* __restrict__ Cf,
                                                      unsigned short* __restrict__ Cb,
                                                      int M, int N, int K) {
    __shared__ __align__(16) unsigned short sA[2][128 * 32];
    __shared__ __align__(16) unsigned short sB[2][128 * 32];

    const int tid   = threadIdx.x;
    const int lane  = tid & 31;
    const int wave  = tid >> 5;
    const int waveM = wave >> 1;          // 0..3
    const int waveN = wave & 1;           // 0..1
    const int half  = lane >> 4;
    const int l15   = lane & 15;
    const int m0    = blockIdx.y * 128;
    const int n0    = blockIdx.x * 128;

    v8f acc[2][4];
    for (int mi = 0; mi < 2; mi++)
        for (int ni = 0; ni < 4; ni++)
            acc[mi][ni] = (v8f){0.f, 0.f, 0.f, 0.f, 0.f, 0.f, 0.f, 0.f};

    // -------- staging bases (hoisted; per stage only +kc*32 elements) --------
    const int crow = tid >> 2;             // 0..63
    const int cseg = (tid & 3) << 3;       // element offset: 0,8,16,24
    const unsigned short* gA0 = A  + (size_t)(m0 + crow) * K + cseg;
    const unsigned short* gA1 = gA0 + (size_t)64 * K;
    const unsigned short* gB0 = Bt + (size_t)(n0 + crow) * K + cseg;
    const unsigned short* gB1 = gB0 + (size_t)64 * K;
    unsigned short* const lA0 = &sA[0][crow * 32 + cseg];
    unsigned short* const lB0 = &sB[0][crow * 32 + cseg];
    const int bufStride = 128 * 32;        // elements per LDS buffer

    auto stage = [&](int buf, int kc) {
        const size_t go = (size_t)kc << 5;
        const int    lo = buf ? bufStride : 0;
        async_copy16(gA0 + go, lA0 + lo);
        async_copy16(gA1 + go, lA0 + lo + 64 * 32);
        async_copy16(gB0 + go, lB0 + lo);
        async_copy16(gB1 + go, lB0 + lo + 64 * 32);
    };

    stage(0, 0);
    const int nkc = K >> 5;
    for (int kc = 0; kc < nkc; kc++) {
        const int cur = kc & 1;
        if (kc + 1 < nkc) {                 // prefetch next tile into other buffer
            stage(cur ^ 1, kc + 1);
            wait_async<4>();                // first 4 (current tile) complete
        } else {
            wait_async<0>();
        }
        __syncthreads();

        const unsigned* pa = (const unsigned*)sA[cur];
        const unsigned* pb = (const unsigned*)sB[cur];
        FragBF fa[2], fb[4];
        for (int mi = 0; mi < 2; mi++) {
            const int row = waveM * 32 + mi * 16 + l15;
            for (int v = 0; v < 8; v++) fa[mi].u[v] = pa[row * 16 + frag_kp(v, half)];
        }
        for (int ni = 0; ni < 4; ni++) {
            const int col = waveN * 64 + ni * 16 + l15;
            for (int v = 0; v < 8; v++) fb[ni].u[v] = pb[col * 16 + frag_kp(v, half)];
        }
        for (int mi = 0; mi < 2; mi++)
            for (int ni = 0; ni < 4; ni++)
                acc[mi][ni] = __builtin_amdgcn_wmma_f32_16x16x32_bf16(
                    false, fa[mi].v, false, fb[ni].v, (short)0, acc[mi][ni], false, false);
        __syncthreads();
    }

    // epilogue: lane holds column l15, rows v + 8*half
    for (int mi = 0; mi < 2; mi++)
        for (int ni = 0; ni < 4; ni++)
            for (int v = 0; v < 8; v++) {
                const int m = m0 + waveM * 32 + mi * 16 + v + 8 * half;
                const int n = n0 + waveN * 64 + ni * 16 + l15;
                const size_t idx = (size_t)m * N + n;
                if (MODE == 2) {
                    Cb[idx] = f2bf(acc[mi][ni][v]);
                } else if (MODE == 1) {
                    Cf[idx] = R[idx] + acc[mi][ni][v];
                } else {
                    Cf[idx] = acc[mi][ni][v];
                }
            }
}

// ---------------------------------------------------------------------------
// RoPE (fp32 in -> bf16 out).  grid = S*nheads blocks of 128 threads.
// ---------------------------------------------------------------------------
__global__ __launch_bounds__(128) void rope_bf16_kernel(const float* __restrict__ x,
                                                        const int* __restrict__ pos,
                                                        unsigned short* __restrict__ out,
                                                        int nheads) {
    const int s = blockIdx.x / nheads;
    const int h = blockIdx.x % nheads;
    const int d = threadIdx.x;                 // 0..127
    const size_t base = ((size_t)s * nheads + h) * HD;
    const float v = x[base + d];
    const int   j = d & 63;
    const float inv_freq = __expf(-logf(1.0e9f) * ((float)(2 * j) / (float)HD));
    const float ang = (float)pos[s] * inv_freq;
    const float c = cosf(ang), sn = sinf(ang);
    const float rot = (d < 64) ? -x[base + d + 64] : x[base + d - 64];
    out[base + d] = f2bf(v * c + rot * sn);
}

// ---------------------------------------------------------------------------
// elementwise fp32 -> bf16
// ---------------------------------------------------------------------------
__global__ __launch_bounds__(256) void cvt_bf16_kernel(const float* __restrict__ x,
                                                       unsigned short* __restrict__ out,
                                                       int n) {
    int i = blockIdx.x * 256 + threadIdx.x;
    if (i < n) out[i] = f2bf(x[i]);
}

// ---------------------------------------------------------------------------
// Flash attention (GQA, causal).  One wave per (head, 16-row query block).
// QK^T and P@V via bf16 WMMA; online softmax owned by 16 row lanes.
// bf16 output (feeds the wo GEMM directly).
// ---------------------------------------------------------------------------
#define ATT_WAVES 4
__global__ __launch_bounds__(128) void attn_kernel(const unsigned short* __restrict__ q,
                                                   const unsigned short* __restrict__ k,
                                                   const unsigned short* __restrict__ vv,
                                                   unsigned short* __restrict__ attnb) {
    const int lane  = threadIdx.x & 31;
    const int wid   = threadIdx.x >> 5;
    const int task  = blockIdx.x * ATT_WAVES + wid;
    const int h     = task >> 7;               // / (2048/16)
    const int qblk  = task & 127;
    const int kvh   = h >> 2;                  // G = 4
    const int half  = lane >> 4;
    const int l15   = lane & 15;

    __shared__ float    lds_s[ATT_WAVES][16 * 32];
    __shared__ unsigned lds_p[ATT_WAVES][16 * 16];
    __shared__ unsigned lds_v[ATT_WAVES][128 * 16];
    __shared__ float    lds_rs[ATT_WAVES][16];
    float*    S_ = lds_s[wid];
    unsigned* P_ = lds_p[wid];
    unsigned* V_ = lds_v[wid];
    float*    RS = lds_rs[wid];

    FragBF aq[4];
    {
        const int qrow = qblk * 16 + l15;
        const unsigned* qp = (const unsigned*)(q + (size_t)qrow * (NH * HD) + h * HD);
        for (int c = 0; c < 4; c++)
            for (int v = 0; v < 8; v++)
                aq[c].u[v] = qp[c * 16 + frag_kp(v, half)];
    }

    v8f accO[8];
    for (int f = 0; f < 8; f++) accO[f] = (v8f){0.f, 0.f, 0.f, 0.f, 0.f, 0.f, 0.f, 0.f};

    float m_i = -3.0e38f, l_i = 0.f;
    const int nk = (((qblk + 1) * 16) + 31) >> 5;

    for (int kt = 0; kt < nk; kt++) {
        // ---- scores: S(16q x 32k) = Q @ K^T --------------------------------
        for (int ni = 0; ni < 2; ni++) {
            const int kcol = (kt << 5) + ni * 16 + l15;
            const unsigned* kp_ = (const unsigned*)(k + (size_t)kcol * KVD + kvh * HD);
            v8f s = (v8f){0.f, 0.f, 0.f, 0.f, 0.f, 0.f, 0.f, 0.f};
            for (int c = 0; c < 4; c++) {
                FragBF bk;
                for (int v = 0; v < 8; v++) bk.u[v] = kp_[c * 16 + frag_kp(v, half)];
                s = __builtin_amdgcn_wmma_f32_16x16x32_bf16(
                    false, aq[c].v, false, bk.v, (short)0, s, false, false);
            }
            const int keyi = (kt << 5) + ni * 16 + l15;
            for (int v = 0; v < 8; v++) {
                const int m  = v + 8 * half;
                const int qi = qblk * 16 + m;
                float val = s[v] * 0.08838834764831845f;   // 1/sqrt(128)
                if (keyi > qi) val = -1.0e9f;
                S_[m * 32 + ni * 16 + l15] = val;
            }
        }
        wave_lds_sync();

        // ---- online softmax (row-owner lanes) ------------------------------
        {
            float pv[32];
            float mc = -3.0e38f;
            for (int j = 0; j < 32; j++) { pv[j] = S_[l15 * 32 + j]; mc = fmaxf(mc, pv[j]); }
            const float nm = fmaxf(m_i, mc);
            const float r  = __expf(m_i - nm);
            float sum = 0.f;
            for (int j = 0; j < 32; j++) { pv[j] = __expf(pv[j] - nm); sum += pv[j]; }
            l_i = l_i * r + sum;
            m_i = nm;
            if (lane < 16) {
                RS[l15] = r;
                for (int kp = 0; kp < 16; kp++)
                    P_[l15 * 16 + kp] = pack2(pv[2 * kp], pv[2 * kp + 1]);
            }
        }
        wave_lds_sync();

        // ---- rescale running O ---------------------------------------------
        for (int f = 0; f < 8; f++)
            for (int v = 0; v < 8; v++)
                accO[f][v] *= RS[v + 8 * half];

        // ---- stage V chunk (32 keys x 128 d) transposed into LDS -----------
        for (int i = lane; i < 128 * 16; i += 32) {
            const int d  = i >> 4, kp = i & 15;
            const int k0 = (kt << 5) + (kp << 1);
            const unsigned short v0 = vv[(size_t)k0 * KVD + kvh * HD + d];
            const unsigned short v1 = vv[(size_t)(k0 + 1) * KVD + kvh * HD + d];
            V_[d * 16 + kp] = (unsigned)v0 | ((unsigned)v1 << 16);
        }
        wave_lds_sync();

        // ---- O += P @ V -----------------------------------------------------
        FragBF ap;
        for (int v = 0; v < 8; v++) ap.u[v] = P_[l15 * 16 + frag_kp(v, half)];
        for (int f = 0; f < 8; f++) {
            FragBF bv;
            for (int v = 0; v < 8; v++)
                bv.u[v] = V_[(f * 16 + l15) * 16 + frag_kp(v, half)];
            accO[f] = __builtin_amdgcn_wmma_f32_16x16x32_bf16(
                false, ap.v, false, bv.v, (short)0, accO[f], false, false);
        }
        wave_lds_sync();   // protect S_/P_/V_ before next tile overwrites
    }

    RS[l15] = 1.0f / l_i;          // lanes 16..31 duplicate identical values
    wave_lds_sync();
    for (int f = 0; f < 8; f++)
        for (int v = 0; v < 8; v++) {
            const int m   = v + 8 * half;
            const int row = qblk * 16 + m;
            attnb[(size_t)row * (NH * HD) + h * HD + f * 16 + l15] =
                f2bf(accO[f][v] * RS[m]);
        }
}

// ---------------------------------------------------------------------------
// SwiGLU combine (bf16 in/out, in place): g = silu(g) * u
// ---------------------------------------------------------------------------
__global__ __launch_bounds__(256) void silumul_kernel(unsigned short* __restrict__ g,
                                                      const unsigned short* __restrict__ u,
                                                      int n) {
    int i = blockIdx.x * 256 + threadIdx.x;
    if (i < n) {
        float x = bf2f(g[i]);
        float y = bf2f(u[i]);
        g[i] = f2bf((x / (1.f + __expf(-x))) * y);
    }
}

// ---------------------------------------------------------------------------
// launch
// ---------------------------------------------------------------------------
extern "C" void kernel_launch(void* const* d_in, const int* in_sizes, int n_in,
                              void* d_out, int out_size, void* d_ws, size_t ws_size,
                              hipStream_t stream) {
    (void)in_sizes; (void)n_in; (void)out_size; (void)ws_size;
    const float* hid    = (const float*)d_in[0];
    const float* wq     = (const float*)d_in[1];
    const float* wk     = (const float*)d_in[2];
    const float* wv     = (const float*)d_in[3];
    const float* wo     = (const float*)d_in[4];
    const float* wgate  = (const float*)d_in[5];
    const float* wup    = (const float*)d_in[6];
    const float* wdown  = (const float*)d_in[7];
    const float* ln1    = (const float*)d_in[8];
    const float* ln2    = (const float*)d_in[9];
    const int*   pos    = (const int*)d_in[10];
    float*       out    = (float*)d_out;

    // ---- workspace layout --------------------------------------------------
    float* qf = (float*)d_ws;                               // [S, HID] f32
    float* kf = qf + (size_t)S_LEN * HID;                   // [S, KVD] f32
    float* vf = kf + (size_t)S_LEN * KVD;                   // [S, KVD] f32
    float* h1 = vf + (size_t)S_LEN * KVD;                   // [S, HID] f32
    unsigned short* xnb = (unsigned short*)(h1 + (size_t)S_LEN * HID);  // [S,HID]
    unsigned short* qb  = xnb + (size_t)S_LEN * HID;        // [S, HID]
    unsigned short* kb  = qb  + (size_t)S_LEN * HID;        // [S, KVD]
    unsigned short* vb  = kb  + (size_t)S_LEN * KVD;        // [S, KVD]
    unsigned short* g16 = vb  + (size_t)S_LEN * KVD;        // [S, FF]
    unsigned short* u16 = g16 + (size_t)S_LEN * FF;         // [S, FF]
    unsigned short* wqT = u16 + (size_t)S_LEN * FF;         // [HID, HID]
    unsigned short* wkT = wqT + (size_t)HID * HID;          // [KVD, HID]
    unsigned short* wvT = wkT + (size_t)KVD * HID;          // [KVD, HID]
    unsigned short* woT = wvT + (size_t)KVD * HID;          // [HID, HID]
    unsigned short* wgT = woT + (size_t)HID * HID;          // [FF, HID]
    unsigned short* wuT = wgT + (size_t)FF * HID;           // [FF, HID]
    unsigned short* wdT = wuT + (size_t)FF * HID;           // [HID, FF]
    unsigned short* attnb = (unsigned short*)kf;            // reuse kf+vf after RoPE

    // ---- weights: fp32 [K,N] -> bf16 [N,K] ----------------------------------
    wcvt_t_kernel<<<dim3(HID / 32, HID / 32), 256, 0, stream>>>(wq, wqT, HID, HID);
    wcvt_t_kernel<<<dim3(KVD / 32, HID / 32), 256, 0, stream>>>(wk, wkT, HID, KVD);
    wcvt_t_kernel<<<dim3(KVD / 32, HID / 32), 256, 0, stream>>>(wv, wvT, HID, KVD);
    wcvt_t_kernel<<<dim3(HID / 32, HID / 32), 256, 0, stream>>>(wo, woT, HID, HID);
    wcvt_t_kernel<<<dim3(FF / 32, HID / 32), 256, 0, stream>>>(wgate, wgT, HID, FF);
    wcvt_t_kernel<<<dim3(FF / 32, HID / 32), 256, 0, stream>>>(wup, wuT, HID, FF);
    wcvt_t_kernel<<<dim3(HID / 32, FF / 32), 256, 0, stream>>>(wdown, wdT, FF, HID);

    // 1) xn1 = rmsnorm(hid, ln1) -> bf16
    rmsnorm_kernel<<<S_LEN, 256, 0, stream>>>(hid, ln1, xnb);

    // 2) q/k/v projections (f32 out for RoPE precision)
    gemm_bf16_wmma<0><<<dim3(HID / 128, S_LEN / 128), 256, 0, stream>>>(xnb, wqT, nullptr, qf, nullptr, S_LEN, HID, HID);
    gemm_bf16_wmma<0><<<dim3(KVD / 128, S_LEN / 128), 256, 0, stream>>>(xnb, wkT, nullptr, kf, nullptr, S_LEN, KVD, HID);
    gemm_bf16_wmma<0><<<dim3(KVD / 128, S_LEN / 128), 256, 0, stream>>>(xnb, wvT, nullptr, vf, nullptr, S_LEN, KVD, HID);

    // 3) RoPE -> bf16 Q/K; bf16 V
    rope_bf16_kernel<<<S_LEN * NH, 128, 0, stream>>>(qf, pos, qb, NH);
    rope_bf16_kernel<<<S_LEN * NKV, 128, 0, stream>>>(kf, pos, kb, NKV);
    {
        int n = S_LEN * KVD;
        cvt_bf16_kernel<<<(n + 255) / 256, 256, 0, stream>>>(vf, vb, n);
    }

    // 4) attention -> bf16 (attnb overlays kf/vf, which are dead now)
    attn_kernel<<<(NH * (S_LEN / 16)) / ATT_WAVES, 32 * ATT_WAVES, 0, stream>>>(qb, kb, vb, attnb);

    // 5) h1 = hid + attn @ wo
    gemm_bf16_wmma<1><<<dim3(HID / 128, S_LEN / 128), 256, 0, stream>>>(attnb, woT, hid, h1, nullptr, S_LEN, HID, HID);

    // 6) xn2 = rmsnorm(h1, ln2) -> bf16
    rmsnorm_kernel<<<S_LEN, 256, 0, stream>>>(h1, ln2, xnb);

    // 7) MLP: gate/up (bf16 out), SwiGLU, down (+residual) -> out
    gemm_bf16_wmma<2><<<dim3(FF / 128, S_LEN / 128), 256, 0, stream>>>(xnb, wgT, nullptr, nullptr, g16, S_LEN, FF, HID);
    gemm_bf16_wmma<2><<<dim3(FF / 128, S_LEN / 128), 256, 0, stream>>>(xnb, wuT, nullptr, nullptr, u16, S_LEN, FF, HID);
    {
        int n = S_LEN * FF;
        silumul_kernel<<<(n + 255) / 256, 256, 0, stream>>>(g16, u16, n);
    }
    gemm_bf16_wmma<1><<<dim3(HID / 128, S_LEN / 128), 256, 0, stream>>>(g16, wdT, h1, out, nullptr, S_LEN, HID, FF);
}
